// MHSA_63067299775190
// MI455X (gfx1250) — matrix operand order repaired and checked by reading, not compiled, not run on hardware
//
#include <hip/hip_runtime.h>

typedef __attribute__((ext_vector_type(16))) __bf16 v16bf;
typedef __attribute__((ext_vector_type(8)))  float  v8f;
typedef unsigned short u16;
typedef unsigned int   u32;
typedef unsigned long long u64;

#define BATCH 16
#define CDIM  256
#define NTOK  2304      // 48*48
#define K2    512       // 2*C contraction for fused logits
#define CH    32        // attention m-chunk
#define BPAD  520       // padded row stride (elts) for B chunk in LDS
#define VPAD  40        // padded row stride (elts) for V/P in LDS

__device__ __forceinline__ u16 f2bf(float x) {
    u32 u = __float_as_uint(x);
    u32 r = u + 0x7FFFu + ((u >> 16) & 1u);   // round-to-nearest-even
    return (u16)(r >> 16);
}

// Raw LDS byte offset (what async-to-LDS VDST expects: LDS_BASE is added by HW)
__device__ __forceinline__ u32 lds_off(const void* p) {
    return (u32)(u64)(__attribute__((address_space(3))) const void*)p;
}

// memory -> LDS, 16B per lane, tracked by ASYNCcnt
__device__ __forceinline__ void async_ld_b128(u32 lds_byte, const void* gaddr) {
    asm volatile("global_load_async_to_lds_b128 %0, %1, off"
                 :: "v"(lds_byte), "v"((u64)gaddr) : "memory");
}
__device__ __forceinline__ void wait_asynccnt0() {
    asm volatile("s_wait_asynccnt 0x0" ::: "memory");
}

union FragBF { v16bf v; uint4 q[2]; };

// ---------------------------------------------------------------------------
// Kernel 0a: transpose+convert x [B,C,N] f32 -> Xt [B,N,C] bf16 (LDS tiled)
// ---------------------------------------------------------------------------
__global__ void transpose_x_kernel(const float* __restrict__ x, u16* __restrict__ Xt) {
    __shared__ float tile[32][33];
    const int tx = threadIdx.x, ty = threadIdx.y;
    const int n0 = blockIdx.x * 32, c0 = blockIdx.y * 32, b = blockIdx.z;
#pragma unroll
    for (int i = 0; i < 4; ++i) {
        int cl = ty + i * 8;
        tile[cl][tx] = x[((size_t)b * CDIM + c0 + cl) * NTOK + n0 + tx];
    }
    __syncthreads();
#pragma unroll
    for (int i = 0; i < 4; ++i) {
        int nl = ty + i * 8;
        Xt[((size_t)b * NTOK + n0 + nl) * CDIM + c0 + tx] = f2bf(tile[tx][nl]);
    }
}

// ---------------------------------------------------------------------------
// Kernel 0b: pos table into Aext[b][n][256+c].  Reference flattens pos as
// [C,W,H] -> n = w*H + h (NOT h*W+w), replicated here exactly.
// ---------------------------------------------------------------------------
__global__ void pos_kernel(const float* __restrict__ rel_h, const float* __restrict__ rel_w,
                           u16* __restrict__ Aext) {
    const int n = blockIdx.x;
    const int c = threadIdx.x;
    const int w = n / 48, h = n % 48;
    u16 v = f2bf(rel_h[c * 48 + h] + rel_w[c * 48 + w]);
#pragma unroll
    for (int b = 0; b < BATCH; ++b)
        Aext[((size_t)b * NTOK + n) * K2 + CDIM + c] = v;
}

// ---------------------------------------------------------------------------
// Kernel 0c: Wcat [768][256] bf16 = concat(Wq,Wk,Wv)
// ---------------------------------------------------------------------------
__global__ void wcat_kernel(const float* __restrict__ Wq, const float* __restrict__ Wk,
                            const float* __restrict__ Wv, u16* __restrict__ Wcat) {
    const int o3 = blockIdx.x, c = threadIdx.x;
    const float* src = (o3 < 256) ? (Wq + (size_t)o3 * 256)
                     : (o3 < 512) ? (Wk + (size_t)(o3 - 256) * 256)
                                  : (Wv + (size_t)(o3 - 512) * 256);
    Wcat[(size_t)o3 * 256 + c] = f2bf(src[c]);
}

// ---------------------------------------------------------------------------
// Kernel 1: projections.  D = Wcat(768x256) x Xt[b]^T + bias, per 16x16 tile
// per wave, K=256 in 8 WMMA steps.  q -> Aext[:, :256] & Bext[:, 256:]
// (transposed writes), k -> Bext[:, :256], v -> Vmat[b][c][m] (natural).
// ---------------------------------------------------------------------------
__global__ __launch_bounds__(256) void proj_kernel(
    const u16* __restrict__ Wcat, const u16* __restrict__ Xt,
    const float* __restrict__ bq, const float* __restrict__ bk, const float* __restrict__ bv,
    u16* __restrict__ Aext, u16* __restrict__ Bext, u16* __restrict__ Vmat) {

    const int tid = threadIdx.x;
    const int wave = tid >> 5, lane = tid & 31;
    const int lhi = lane >> 4, llo = lane & 15;
    const int b = blockIdx.y;
    const int wid = blockIdx.x * 8 + wave;        // 0..6911
    const int mt = wid % 48, nt = wid / 48;
    const int o3 = mt * 16, nb = nt * 16;

    const u16* Xb = Xt + (size_t)b * NTOK * CDIM;

    v8f acc = {};
#pragma unroll
    for (int k = 0; k < CDIM; k += 32) {
        FragBF fa, fb;
        const u16* ar = Wcat + (size_t)(o3 + llo) * CDIM + k + 8 * lhi;
        fa.q[0] = *(const uint4*)(ar);
        fa.q[1] = *(const uint4*)(ar + 16);
        const u16* br = Xb + (size_t)(nb + llo) * CDIM + k + 8 * lhi;
        fb.q[0] = *(const uint4*)(br);
        fb.q[1] = *(const uint4*)(br + 16);
        acc = __builtin_amdgcn_wmma_f32_16x16x32_bf16(false, fa.v, false, fb.v,
                                                      (short)0, acc, false, false);
    }

    const float* bp = (o3 < 256) ? bq : (o3 < 512 ? bk : bv);
    const int ob = (o3 & 255) + 8 * lhi;
#pragma unroll
    for (int g = 0; g < 8; ++g) acc[g] += bp[ob + g];

    const int n = nb + llo;
    if (o3 < 512) {
        // transposed write: per lane, 8 consecutive output channels of row n
        u32 pk0 = (u32)f2bf(acc[0]) | ((u32)f2bf(acc[1]) << 16);
        u32 pk1 = (u32)f2bf(acc[2]) | ((u32)f2bf(acc[3]) << 16);
        u32 pk2 = (u32)f2bf(acc[4]) | ((u32)f2bf(acc[5]) << 16);
        u32 pk3 = (u32)f2bf(acc[6]) | ((u32)f2bf(acc[7]) << 16);
        uint4 q = make_uint4(pk0, pk1, pk2, pk3);
        if (o3 < 256) {
            *(uint4*)(Aext + ((size_t)b * NTOK + n) * K2 + ob) = q;
            *(uint4*)(Bext + ((size_t)b * NTOK + n) * K2 + CDIM + ob) = q;
        } else {
            *(uint4*)(Bext + ((size_t)b * NTOK + n) * K2 + ob) = q;
        }
    } else {
#pragma unroll
        for (int g = 0; g < 8; ++g) {
            int o = (o3 - 512) + g + 8 * lhi;
            Vmat[((size_t)b * CDIM + o) * NTOK + n] = f2bf(acc[g]);
        }
    }
}

// ---------------------------------------------------------------------------
// Kernel 2: flash attention.  WG = 8 waves x 16 rows = 128 rows of one batch.
// m-chunks of 32 staged into LDS with GLOBAL_LOAD_ASYNC_TO_LDS_B128 (no VGPR
// round trip, ASYNCcnt-tracked), online softmax, O in f32 (16 v8f per wave).
// ---------------------------------------------------------------------------
__global__ __launch_bounds__(256) void attn_kernel(
    const u16* __restrict__ Aext, const u16* __restrict__ Bext,
    const u16* __restrict__ Vmat, float* __restrict__ out) {

    __shared__ u16 Blds[CH * BPAD];        // 33280 B
    __shared__ u16 Vlds[CDIM * VPAD];      // 20480 B
    __shared__ u16 Plds[8 * 16 * VPAD];    // 10240 B  (per-wave P transpose)

    const int tid = threadIdx.x;
    const int wave = tid >> 5, lane = tid & 31;
    const int lhi = lane >> 4, llo = lane & 15;
    const int b = blockIdx.y;
    const int n0 = blockIdx.x * 128 + wave * 16;

    const u16* Ab = Aext + (size_t)b * NTOK * K2;
    const u16* Bb = Bext + (size_t)b * NTOK * K2;
    const u16* Vb = Vmat + (size_t)b * CDIM * NTOK;
    u16* Pw = Plds + wave * (16 * VPAD);

    const u32 BldsBase = lds_off(Blds);
    const u32 VldsBase = lds_off(Vlds);

    const v8f zero = {};
    v8f acc[16];
#pragma unroll
    for (int t = 0; t < 16; ++t) acc[t] = zero;
    v8f mrun, lrun;
#pragma unroll
    for (int g = 0; g < 8; ++g) { mrun[g] = -3.0e38f; lrun[g] = 0.0f; }

    for (int mc = 0; mc < NTOK; mc += CH) {
        __syncthreads();   // previous chunk's LDS reads complete
        {
            // stage Bext[mc:mc+32][0:512] (32KB contiguous) -> padded LDS rows
            const u16* gB = Bb + (size_t)mc * K2;
#pragma unroll
            for (int i = 0; i < 8; ++i) {
                int idx = tid + 256 * i;                 // 2048 x 16B
                int row = idx >> 6, seg = idx & 63;
                async_ld_b128(BldsBase + (u32)(row * BPAD + seg * 8) * 2u,
                              gB + (size_t)idx * 8);
            }
            // stage V[0:256][mc:mc+32] -> padded LDS rows
#pragma unroll
            for (int i = 0; i < 4; ++i) {
                int idx = tid + 256 * i;                 // 1024 x 16B
                int c = idx >> 2, seg = idx & 3;
                async_ld_b128(VldsBase + (u32)(c * VPAD + seg * 8) * 2u,
                              Vb + (size_t)c * NTOK + mc + seg * 8);
            }
            wait_asynccnt0();
        }
        __syncthreads();   // LDS tiles visible to all waves

        // ---- logits: S[16 x 32] over K=512 ----
        v8f s0 = zero, s1 = zero;
#pragma unroll
        for (int k = 0; k < K2; k += 32) {
            FragBF fa, fb0, fb1;
            const u16* ar = Ab + (size_t)(n0 + llo) * K2 + k + 8 * lhi;
            fa.q[0] = *(const uint4*)(ar);
            fa.q[1] = *(const uint4*)(ar + 16);
            const u16* b0 = &Blds[llo * BPAD + k + 8 * lhi];
            fb0.q[0] = *(const uint4*)(b0);
            fb0.q[1] = *(const uint4*)(b0 + 16);
            const u16* b1 = &Blds[(16 + llo) * BPAD + k + 8 * lhi];
            fb1.q[0] = *(const uint4*)(b1);
            fb1.q[1] = *(const uint4*)(b1 + 16);
            s0 = __builtin_amdgcn_wmma_f32_16x16x32_bf16(false, fa.v, false, fb0.v,
                                                         (short)0, s0, false, false);
            s1 = __builtin_amdgcn_wmma_f32_16x16x32_bf16(false, fa.v, false, fb1.v,
                                                         (short)0, s1, false, false);
        }

        // ---- online softmax (rows live across one 16-lane half) ----
        v8f cm;
#pragma unroll
        for (int g = 0; g < 8; ++g) cm[g] = fmaxf(s0[g], s1[g]);
#pragma unroll
        for (int off = 8; off >= 1; off >>= 1)
#pragma unroll
            for (int g = 0; g < 8; ++g) cm[g] = fmaxf(cm[g], __shfl_xor(cm[g], off, 32));

        v8f mnew, alpha, p0, p1, rs;
#pragma unroll
        for (int g = 0; g < 8; ++g) {
            mnew[g]  = fmaxf(mrun[g], cm[g]);
            alpha[g] = __expf(mrun[g] - mnew[g]);
            p0[g]    = __expf(s0[g] - mnew[g]);
            p1[g]    = __expf(s1[g] - mnew[g]);
            rs[g]    = p0[g] + p1[g];
        }
#pragma unroll
        for (int off = 8; off >= 1; off >>= 1)
#pragma unroll
            for (int g = 0; g < 8; ++g) rs[g] += __shfl_xor(rs[g], off, 32);
#pragma unroll
        for (int g = 0; g < 8; ++g) {
            lrun[g] = lrun[g] * alpha[g] + rs[g];
            mrun[g] = mnew[g];
        }
#pragma unroll
        for (int t = 0; t < 16; ++t)
#pragma unroll
            for (int g = 0; g < 8; ++g) acc[t][g] *= alpha[g];

        // ---- D-layout P -> LDS row-major [16 n][32 m] (per-wave, no barrier) ----
#pragma unroll
        for (int g = 0; g < 8; ++g) {
            int prow = g + 8 * lhi;
            Pw[prow * VPAD + llo]      = f2bf(p0[g]);
            Pw[prow * VPAD + 16 + llo] = f2bf(p1[g]);
        }
        FragBF fp;
        {
            const u16* pr = &Pw[llo * VPAD + 8 * lhi];
            fp.q[0] = *(const uint4*)(pr);
            fp.q[1] = *(const uint4*)(pr + 16);
        }
        // ---- O += P x V^T over all 16 channel tiles ----
#pragma unroll
        for (int t = 0; t < 16; ++t) {
            FragBF fv;
            const u16* vr = &Vlds[(t * 16 + llo) * VPAD + 8 * lhi];
            fv.q[0] = *(const uint4*)(vr);
            fv.q[1] = *(const uint4*)(vr + 16);
            acc[t] = __builtin_amdgcn_wmma_f32_16x16x32_bf16(false, fp.v, false, fv.v,
                                                             (short)0, acc[t], false, false);
        }
    }

    // ---- finalize: divide by row sums, write out[b][c][n] f32 ----
    v8f inv;
#pragma unroll
    for (int g = 0; g < 8; ++g) inv[g] = 1.0f / lrun[g];
#pragma unroll
    for (int t = 0; t < 16; ++t) {
        int c = t * 16 + llo;
        float* op = out + ((size_t)b * CDIM + c) * NTOK + n0 + 8 * lhi;
#pragma unroll
        for (int g = 0; g < 8; ++g) op[g] = acc[t][g] * inv[g];
    }
}

// ---------------------------------------------------------------------------
extern "C" void kernel_launch(void* const* d_in, const int* in_sizes, int n_in,
                              void* d_out, int out_size, void* d_ws, size_t ws_size,
                              hipStream_t stream) {
    const float* x     = (const float*)d_in[0];
    const float* Wq    = (const float*)d_in[1];
    const float* bq    = (const float*)d_in[2];
    const float* Wk    = (const float*)d_in[3];
    const float* bk    = (const float*)d_in[4];
    const float* Wv    = (const float*)d_in[5];
    const float* bv    = (const float*)d_in[6];
    const float* rel_h = (const float*)d_in[7];
    const float* rel_w = (const float*)d_in[8];
    float* out = (float*)d_out;

    u16* ws   = (u16*)d_ws;
    u16* Wcat = ws;                                      //  768*256
    u16* Xt   = Wcat + (size_t)768 * 256;                //  B*N*C
    u16* Aext = Xt   + (size_t)BATCH * NTOK * CDIM;      //  B*N*512  [Q | pos]
    u16* Bext = Aext + (size_t)BATCH * NTOK * K2;        //  B*N*512  [K | Q]
    u16* Vmat = Bext + (size_t)BATCH * NTOK * K2;        //  B*C*N

    wcat_kernel<<<dim3(768), dim3(256), 0, stream>>>(Wq, Wk, Wv, Wcat);
    transpose_x_kernel<<<dim3(NTOK / 32, CDIM / 32, BATCH), dim3(32, 8), 0, stream>>>(x, Xt);
    pos_kernel<<<dim3(NTOK), dim3(256), 0, stream>>>(rel_h, rel_w, Aext);
    proj_kernel<<<dim3((48 * 144) / 8, BATCH), dim3(256), 0, stream>>>(
        Wcat, Xt, bq, bk, bv, Aext, Bext, Vmat);
    attn_kernel<<<dim3(NTOK / 128, BATCH), dim3(256), 0, stream>>>(Aext, Bext, Vmat, out);
}
